// GptOssBlock_38963943309759
// MI455X (gfx1250) — matrix operand order repaired and checked by reading, not compiled, not run on hardware
//
#include <hip/hip_runtime.h>
#include <hip/hip_bf16.h>
#include <math.h>

// ---------------- CDNA5 WMMA/TDM types/helpers ----------------
typedef __attribute__((ext_vector_type(16))) _Float16 v16h;
typedef __attribute__((ext_vector_type(8)))  _Float16 v8h;
typedef __attribute__((ext_vector_type(8)))  float    v8f;
typedef __attribute__((ext_vector_type(4)))  unsigned int v4u;
typedef __attribute__((ext_vector_type(8)))  int      v8i;
typedef __attribute__((ext_vector_type(4)))  int      v4i;

#define NH   32
#define KVH  8
#define HD   64
#define SEQ  1024
#define DIM  2048
#define HID  2048
#define NEXP 16
#define TOPK 4

__device__ __forceinline__ v8f wmma_f16(v16h a, v16h b, v8f c) {
  // D = A(16x32 f16) * B(32x16 f16) + C(16x16 f32)
  return __builtin_amdgcn_wmma_f32_16x16x32_f16(false, a, false, b, (short)0, c,
                                                false, false);
}

__device__ __forceinline__ v16h cat8(v8h lo, v8h hi) {
  return __builtin_shufflevector(lo, hi, 0, 1, 2, 3, 4, 5, 6, 7,
                                 8, 9, 10, 11, 12, 13, 14, 15);
}
// 16 halves from two 16B chunks at p and p+16 halves (A-fragment pattern).
__device__ __forceinline__ v16h ld16_split(const _Float16* p) {
  return cat8(*(const v8h*)p, *(const v8h*)(p + 16));
}
// 16 contiguous halves (packed-B pattern), 16B aligned.
__device__ __forceinline__ v16h ld16_adj(const _Float16* p) {
  const v8h* q = (const v8h*)p;
  return cat8(q[0], q[1]);
}
// A fragment from row-major f16: lane l -> M=l&15, K=8*(l>=16)+i (+16 for hi 8)
__device__ __forceinline__ v16h load_a16(const _Float16* __restrict__ A, int lda,
                                         int row, int k0, int lane) {
  const int kb = (lane >> 4) << 3;
  return ld16_split(A + (size_t)row * lda + k0 + kb);
}

// TDM: stage a 16-row x K-halves panel of row-major f16 into LDS (2D tile).
// D# per CDNA5 ISA 8.3/8.4: group0={count,lds,gaddr,type}, group1={dsize,dims}.
// This toolchain's builtin takes 6 args: (g0 v4u, g1 v8i, g2 v4i, g3 v4i, v8i, cpol)
__device__ __forceinline__ void tdm_load_panel(const _Float16* gsrc,
                                               _Float16* lds_dst, int K) {
  const unsigned long long ga = (unsigned long long)(uintptr_t)gsrc;
  const unsigned int lds = (unsigned int)(uintptr_t)lds_dst;  // LDS byte offset
  const unsigned int dim0 = (unsigned int)K;                  // elements per row
  const unsigned long long str0 = (unsigned long long)K;      // row stride (elems)
  v4u g0 = {};
  g0[0] = 1u;                                            // count=1, user mode
  g0[1] = lds;                                           // lds_addr
  g0[2] = (unsigned int)ga;                              // global_addr[31:0]
  g0[3] = (unsigned int)((ga >> 32) & 0x1FFFFFFu) | (2u << 30);  // [56:32]|type=2
  v8i g1 = {};
  g1[0] = (int)(1u << 16);                               // data_size = 2 bytes
  g1[1] = (int)((dim0 & 0xFFFFu) << 16);                 // tensor_dim0[15:0]
  g1[2] = (int)((dim0 >> 16) | (16u << 16));             // dim0[31:16] | dim1=16
  g1[3] = (int)((dim0 & 0xFFFFu) << 16);                 // dim1 hi=0 | tile_dim0=K
  g1[4] = 16;                                            // tile_dim1=16, tile_dim2=0
  g1[5] = (int)(str0 & 0xFFFFFFFFu);                     // tensor_dim0_stride lo
  g1[6] = (int)((str0 >> 32) & 0xFFFFu);                 // stride hi | dim1_stride=0
  g1[7] = 0;
  v4i gz = {};
  v8i gz8 = {};
  __builtin_amdgcn_tensor_load_to_lds(g0, g1, gz, gz, gz8, 0);
  __builtin_amdgcn_s_wait_tensorcnt(0);
}

// ---------------- Weight pre-pack: f32 [K][N] -> f16 WMMA-B fragment order ----
// NOTE: all call sites have K == 2048 (ktc = 64) -> shift/mask tile decode.
__global__ void k_pack(const float* __restrict__ W, _Float16* __restrict__ Wp,
                       int K, int N) {
  const size_t mat = blockIdx.y;                     // expert index (or 0)
  const float* Wm = W + mat * (size_t)K * N;
  _Float16* Wpm = Wp + mat * (size_t)K * N;
  const int chunk = blockIdx.x * blockDim.x + threadIdx.x;  // 8 halves per thread
  const int o = chunk << 3;
  if (o >= K * N) return;
  const int T = o >> 9;
  const int r = o & 511;
  const int l = r >> 4;                              // lane
  const int i0 = r & 15;                             // 0 or 8
  const int nt = T >> 6, kt = T & 63;                // ktc == 64
  const int n = nt * 16 + (l & 15);
  const int kbase = kt * 32 + ((l >> 4) << 3) + (i0 << 1);
  v8h out;
#pragma unroll
  for (int j = 0; j < 8; ++j) out[j] = (_Float16)Wm[(size_t)(kbase + j) * N + n];
  *(v8h*)(Wpm + o) = out;
}

// ---------------- RMSNorm ----------------
__global__ void k_rmsnorm(const float* __restrict__ x, const float* __restrict__ w,
                          float* __restrict__ of, _Float16* __restrict__ oh) {
  const int t = blockIdx.x, tid = threadIdx.x;
  __shared__ float red[256];
  float s = 0.f;
  for (int d = tid; d < DIM; d += 256) { float v = x[(size_t)t * DIM + d]; s += v * v; }
  red[tid] = s; __syncthreads();
  for (int off = 128; off >= 1; off >>= 1) {
    if (tid < off) red[tid] += red[tid + off];
    __syncthreads();
  }
  const float inv = rsqrtf(red[0] / (float)DIM + 1e-5f);
  for (int d = tid; d < DIM; d += 256) {
    float v = x[(size_t)t * DIM + d] * inv * w[d];
    if (of) of[(size_t)t * DIM + d] = v;
    oh[(size_t)t * DIM + d] = (_Float16)v;
  }
}

// -------- Generic GEMM: C = A_f16[MxK] * Bp_f16(packed)[KxN] (+res) --------
// A panel staged via TDM into LDS; requires K == 2048 at all call sites.
__global__ __launch_bounds__(128) void k_gemm(const _Float16* __restrict__ A,
                                              const _Float16* __restrict__ Bp,
                                              const float* __restrict__ res,
                                              float* __restrict__ Cf,
                                              _Float16* __restrict__ Ch,
                                              int M, int N, int K) {
  __shared__ __align__(16) _Float16 atile[16 * 2048];
  const int lane = threadIdx.x;
  const int row0 = blockIdx.x * 16;
  const int col0 = (blockIdx.y * blockDim.y + threadIdx.y) * 64;
  if (threadIdx.y == 0)
    tdm_load_panel(A + (size_t)row0 * K, atile, K);   // one TDM op per block
  __syncthreads();
  if (col0 >= N || row0 >= M) return;
  const int ktc = K >> 5, nt0 = col0 >> 4;
  const _Float16* b0 = Bp + (((size_t)(nt0 + 0) * ktc) << 9) + lane * 16;
  const _Float16* b1 = Bp + (((size_t)(nt0 + 1) * ktc) << 9) + lane * 16;
  const _Float16* b2 = Bp + (((size_t)(nt0 + 2) * ktc) << 9) + lane * 16;
  const _Float16* b3 = Bp + (((size_t)(nt0 + 3) * ktc) << 9) + lane * 16;
  const _Float16* ap = atile + (size_t)(lane & 15) * K + ((lane >> 4) << 3);
  v8f acc[4] = {};
  for (int k0 = 0; k0 < K; k0 += 32) {
    v16h a = ld16_split(ap); ap += 32;
    acc[0] = wmma_f16(a, ld16_adj(b0), acc[0]); b0 += 512;
    acc[1] = wmma_f16(a, ld16_adj(b1), acc[1]); b1 += 512;
    acc[2] = wmma_f16(a, ld16_adj(b2), acc[2]); b2 += 512;
    acc[3] = wmma_f16(a, ld16_adj(b3), acc[3]); b3 += 512;
  }
  const int half_sel = lane >> 4, col_l = lane & 15;
#pragma unroll
  for (int t = 0; t < 4; ++t)
#pragma unroll
    for (int g = 0; g < 8; ++g) {
      const int r = row0 + g + 8 * half_sel;
      const int c = col0 + t * 16 + col_l;
      float v = acc[t][g];
      if (res) v += res[(size_t)r * N + c];
      if (Cf)  Cf[(size_t)r * N + c] = v;
      if (Ch)  Ch[(size_t)r * N + c] = (_Float16)v;
    }
}

// ---------------- RoPE (f32 -> f16) ----------------
__global__ void k_rope(const float* __restrict__ src, _Float16* __restrict__ dst,
                       int nh, int total) {
  const int gid = blockIdx.x * blockDim.x + threadIdx.x;
  if (gid >= total) return;
  const int d = gid & 31;
  const int rem = gid >> 5;
  const int h = rem % nh;
  const int t = rem / nh;
  const float inv = powf(150000.0f, -(float)d / 32.0f);
  const float ang = (float)t * inv;
  const float c = cosf(ang), sn = sinf(ang);
  const size_t base = ((size_t)t * nh + h) * HD;
  const float x1 = src[base + d], x2 = src[base + d + 32];
  dst[base + d]      = (_Float16)(x1 * c - x2 * sn);
  dst[base + d + 32] = (_Float16)(x2 * c + x1 * sn);
}

// V: f32 [t][kvh][d] -> f16 packed [kvh][d][t] so PV B-fragments are contiguous
__global__ void k_pack_v(const float* __restrict__ vf, _Float16* __restrict__ vp) {
  const int gid = blockIdx.x * blockDim.x + threadIdx.x;
  if (gid >= SEQ * KVH * HD) return;
  const int t = gid >> 9;          // / (KVH*HD)
  const int r = gid & 511;
  vp[(size_t)r * SEQ + t] = (_Float16)vf[gid];   // r = kvh*64 + d
}

// ---------------- Flash attention: one wave per (head, 16-query tile) ----------
__global__ __launch_bounds__(32) void k_attention(const _Float16* __restrict__ qh,
                                                  const _Float16* __restrict__ kh,
                                                  const _Float16* __restrict__ vp,
                                                  _Float16* __restrict__ oh) {
  const int lane = threadIdx.x;
  const int qtile = blockIdx.x;       // 0..63
  const int head  = blockIdx.y;       // 0..31
  const int kvh   = head >> 2;        // GQA group of 4
  __shared__ __align__(16) _Float16 pbuf[16 * 32];

  v16h qa[2];
  {
    const int mrow = qtile * 16 + (lane & 15);
    const int kb = (lane >> 4) << 3;
    const _Float16* qrow = qh + ((size_t)mrow * NH + head) * HD;
#pragma unroll
    for (int step = 0; step < 2; ++step) qa[step] = ld16_split(qrow + step * 32 + kb);
  }

  v8f oacc[4] = {};
  float mrow[8], lrow[8];
#pragma unroll
  for (int g = 0; g < 8; ++g) { mrow[g] = -1e30f; lrow[g] = 0.f; }
  const int half_sel = lane >> 4, col_l = lane & 15;
  const int qmax = qtile * 16 + 15;
  const int nkt = (qmax / 32) + 1;

  for (int kt = 0; kt < nkt; ++kt) {
    v8f sacc[2] = {};
#pragma unroll
    for (int sub = 0; sub < 2; ++sub) {
      const int key = kt * 32 + sub * 16 + col_l;
#pragma unroll
      for (int step = 0; step < 2; ++step) {
        const int d0 = step * 32 + (half_sel << 3);
        v16h b = ld16_split(kh + ((size_t)key * KVH + kvh) * HD + d0);
        sacc[sub] = wmma_f16(qa[step], b, sacc[sub]);
      }
    }
    float mnew[8];
#pragma unroll
    for (int sub = 0; sub < 2; ++sub)
#pragma unroll
      for (int g = 0; g < 8; ++g) {
        const int qi = qtile * 16 + g + 8 * half_sel;
        const int ki = kt * 32 + sub * 16 + col_l;
        float v = sacc[sub][g] * 0.125f;
        if (ki > qi) v = -1e30f;
        sacc[sub][g] = v;
      }
#pragma unroll
    for (int g = 0; g < 8; ++g) {
      float v = fmaxf(sacc[0][g], sacc[1][g]);
      for (int off = 8; off >= 1; off >>= 1) v = fmaxf(v, __shfl_xor(v, off, 32));
      mnew[g] = fmaxf(mrow[g], v);
    }
    float rsum[8];
#pragma unroll
    for (int g = 0; g < 8; ++g) rsum[g] = 0.f;
#pragma unroll
    for (int sub = 0; sub < 2; ++sub)
#pragma unroll
      for (int g = 0; g < 8; ++g) {
        float p = __expf(sacc[sub][g] - mnew[g]);
        sacc[sub][g] = p;
        rsum[g] += p;
      }
#pragma unroll
    for (int g = 0; g < 8; ++g) {
      float v = rsum[g];
      for (int off = 8; off >= 1; off >>= 1) v += __shfl_xor(v, off, 32);
      const float corr = __expf(mrow[g] - mnew[g]);
      lrow[g] = lrow[g] * corr + v;
      mrow[g] = mnew[g];
#pragma unroll
      for (int t = 0; t < 4; ++t) oacc[t][g] *= corr;
    }
    __syncthreads();
#pragma unroll
    for (int sub = 0; sub < 2; ++sub)
#pragma unroll
      for (int g = 0; g < 8; ++g)
        pbuf[(g + 8 * half_sel) * 32 + sub * 16 + col_l] = (_Float16)sacc[sub][g];
    __syncthreads();
    v16h pa = ld16_split(pbuf + (lane & 15) * 32 + ((lane >> 4) << 3));
#pragma unroll
    for (int t = 0; t < 4; ++t) {
      const int d = t * 16 + col_l;
      const int kkb = half_sel << 3;
      v16h b = ld16_split(vp + ((size_t)(kvh * HD + d)) * SEQ + kt * 32 + kkb);
      oacc[t] = wmma_f16(pa, b, oacc[t]);
    }
  }
#pragma unroll
  for (int t = 0; t < 4; ++t)
#pragma unroll
    for (int g = 0; g < 8; ++g) {
      const int r = qtile * 16 + g + 8 * half_sel;
      const int d = t * 16 + col_l;
      oh[(size_t)r * (NH * HD) + head * HD + d] = (_Float16)(oacc[t][g] / lrow[g]);
    }
}

// ---------------- MoE gate + routing ----------------
__global__ void k_zero_counts(int* counts) {
  if (threadIdx.x < NEXP) counts[threadIdx.x] = 0;
}

__global__ void k_gate(const float* __restrict__ h2, const float* __restrict__ wg,
                       float* __restrict__ gate_w, int* __restrict__ counts,
                       int* __restrict__ bucket) {
  const int t = blockIdx.x, tid = threadIdx.x;
  const int e = tid >> 4, ln = tid & 15;
  __shared__ float red[256];
  float s = 0.f;
  for (int d = ln; d < DIM; d += 16) s += h2[(size_t)t * DIM + d] * wg[d * NEXP + e];
  red[tid] = s; __syncthreads();
  for (int off = 8; off >= 1; off >>= 1) {
    if (ln < off) red[tid] += red[tid + off];
    __syncthreads();
  }
  if (tid == 0) {
    float logit[NEXP];
    for (int i = 0; i < NEXP; ++i) logit[i] = red[i * 16];
    int ti[TOPK]; float tv[TOPK]; bool used[NEXP] = {};
    for (int s2 = 0; s2 < TOPK; ++s2) {
      float best = -3.4e38f; int bi = 0;
      for (int i = 0; i < NEXP; ++i)
        if (!used[i] && logit[i] > best) { best = logit[i]; bi = i; }
      used[bi] = true; ti[s2] = bi; tv[s2] = best;
    }
    const float mx = tv[0];
    float se = 0.f, g[TOPK];
    for (int s2 = 0; s2 < TOPK; ++s2) { g[s2] = __expf(tv[s2] - mx); se += g[s2]; }
    for (int i = 0; i < NEXP; ++i) gate_w[t * NEXP + i] = 0.f;
    for (int s2 = 0; s2 < TOPK; ++s2) {
      gate_w[t * NEXP + ti[s2]] = g[s2] / se;
      const int pos = atomicAdd(&counts[ti[s2]], 1);
      bucket[ti[s2] * SEQ + pos] = t | (s2 << 16);
    }
  }
}

__global__ void k_scan(const int* __restrict__ counts, int* __restrict__ offsets) {
  if (threadIdx.x == 0 && blockIdx.x == 0) {
    int o = 0;
    for (int e = 0; e < NEXP; ++e) { offsets[e] = o; o += counts[e]; }
    offsets[NEXP] = o;
  }
}

// ---------------- MoE stage 1: act = silu(h*w1[e]) * (h*w3[e]) ----------------
__global__ __launch_bounds__(128) void k_moe1(const _Float16* __restrict__ h2h,
                                              const _Float16* __restrict__ w1p,
                                              const _Float16* __restrict__ w3p,
                                              const int* __restrict__ counts,
                                              const int* __restrict__ offsets,
                                              const int* __restrict__ bucket,
                                              _Float16* __restrict__ act) {
  const int e = blockIdx.x, tile = blockIdx.y;
  const int cnt = counts[e];
  if (tile * 16 >= cnt) return;
  const int lane = threadIdx.x;
  const int col0 = (blockIdx.z * blockDim.y + threadIdx.y) * 64;
  int idxA = tile * 16 + (lane & 15);
  if (idxA >= cnt) idxA = cnt - 1;
  const int token = bucket[e * SEQ + idxA] & 0xFFFF;
  const int ktc = DIM >> 5, nt0 = col0 >> 4;
  const _Float16* base1 = w1p + (size_t)e * DIM * HID + lane * 16;
  const _Float16* base3 = w3p + (size_t)e * DIM * HID + lane * 16;
  const _Float16* b1p[4];
  const _Float16* b3p[4];
#pragma unroll
  for (int t = 0; t < 4; ++t) {
    b1p[t] = base1 + (((size_t)(nt0 + t) * ktc) << 9);
    b3p[t] = base3 + (((size_t)(nt0 + t) * ktc) << 9);
  }
  const _Float16* ap = h2h + (size_t)token * DIM + ((lane >> 4) << 3);
  v8f a1[4] = {}, a3[4] = {};
  for (int k0 = 0; k0 < DIM; k0 += 32) {
    v16h a = ld16_split(ap); ap += 32;
#pragma unroll
    for (int t = 0; t < 4; ++t) {
      a1[t] = wmma_f16(a, ld16_adj(b1p[t]), a1[t]); b1p[t] += 512;
      a3[t] = wmma_f16(a, ld16_adj(b3p[t]), a3[t]); b3p[t] += 512;
    }
  }
  const int ro = offsets[e], half_sel = lane >> 4, col_l = lane & 15;
#pragma unroll
  for (int t = 0; t < 4; ++t)
#pragma unroll
    for (int g = 0; g < 8; ++g) {
      const int irow = tile * 16 + g + 8 * half_sel;
      if (irow < cnt) {
        const float g1 = a1[t][g], g3v = a3[t][g];
        const float av = (g1 / (1.0f + __expf(-g1))) * g3v;  // silu(g1)*g3
        act[(size_t)(ro + irow) * HID + col0 + t * 16 + col_l] = (_Float16)av;
      }
    }
}

// ---------------- MoE stage 2: moe_y[t*4+s] = gate * (act * w2[e]) ----------------
__global__ __launch_bounds__(128) void k_moe2(const _Float16* __restrict__ act,
                                              const _Float16* __restrict__ w2p,
                                              const float* __restrict__ gate_w,
                                              const int* __restrict__ counts,
                                              const int* __restrict__ offsets,
                                              const int* __restrict__ bucket,
                                              float* __restrict__ moe_y) {
  const int e = blockIdx.x, tile = blockIdx.y;
  const int cnt = counts[e];
  if (tile * 16 >= cnt) return;
  const int lane = threadIdx.x;
  const int col0 = (blockIdx.z * blockDim.y + threadIdx.y) * 64;
  int idxA = tile * 16 + (lane & 15);
  if (idxA >= cnt) idxA = cnt - 1;
  const int rowA = offsets[e] + idxA;
  const int ktc = HID >> 5, nt0 = col0 >> 4;
  const _Float16* baseB = w2p + (size_t)e * HID * DIM + lane * 16;
  const _Float16* bp[4];
#pragma unroll
  for (int t = 0; t < 4; ++t) bp[t] = baseB + (((size_t)(nt0 + t) * ktc) << 9);
  const _Float16* ap = act + (size_t)rowA * HID + ((lane >> 4) << 3);
  v8f acc[4] = {};
  for (int k0 = 0; k0 < HID; k0 += 32) {
    v16h a = ld16_split(ap); ap += 32;
#pragma unroll
    for (int t = 0; t < 4; ++t) {
      acc[t] = wmma_f16(a, ld16_adj(bp[t]), acc[t]); bp[t] += 512;
    }
  }
  const int half_sel = lane >> 4, col_l = lane & 15;
#pragma unroll
  for (int t = 0; t < 4; ++t)
#pragma unroll
    for (int g = 0; g < 8; ++g) {
      const int irow = tile * 16 + g + 8 * half_sel;
      if (irow < cnt) {
        const int entry = bucket[e * SEQ + irow];
        const int token = entry & 0xFFFF;
        const int slot  = (entry >> 16) & 3;
        const float wgt = gate_w[token * NEXP + e];
        moe_y[((size_t)token * TOPK + slot) * DIM + col0 + t * 16 + col_l] =
            acc[t][g] * wgt;
      }
    }
}

// ---------------- Final combine: out = x1 + sum_slots moe_y ----------------
__global__ void k_combine(const float* __restrict__ x1, const float* __restrict__ moe_y,
                          float* __restrict__ out) {
  const int gid = blockIdx.x * blockDim.x + threadIdx.x;
  if (gid >= SEQ * DIM) return;
  const int t = gid >> 11, d = gid & (DIM - 1);
  float s = x1[gid];
#pragma unroll
  for (int sl = 0; sl < TOPK; ++sl) s += moe_y[((size_t)t * TOPK + sl) * DIM + d];
  out[gid] = s;
}

// ---------------- host launch ----------------
extern "C" void kernel_launch(void* const* d_in, const int* in_sizes, int n_in,
                              void* d_out, int out_size, void* d_ws, size_t ws_size,
                              hipStream_t stream) {
  const float* x    = (const float*)d_in[0];
  const float* ln_w = (const float*)d_in[1];
  const float* ff_w = (const float*)d_in[2];
  const float* wq   = (const float*)d_in[3];
  const float* wk   = (const float*)d_in[4];
  const float* wv   = (const float*)d_in[5];
  const float* wo   = (const float*)d_in[6];
  const float* wg   = (const float*)d_in[7];
  const float* w1   = (const float*)d_in[8];
  const float* w3   = (const float*)d_in[9];
  const float* w2   = (const float*)d_in[10];
  float* out = (float*)d_out;

  char* ws = (char*)d_ws;
  size_t off = 0;
  auto alloc = [&](size_t bytes) {
    char* p = ws + off;
    off += (bytes + 255) & ~(size_t)255;
    return p;
  };
  _Float16* h1h  = (_Float16*)alloc((size_t)SEQ * DIM * 2);
  float*    qf   = (float*)   alloc((size_t)SEQ * NH * HD * 4);
  float*    kf   = (float*)   alloc((size_t)SEQ * KVH * HD * 4);
  float*    vf   = (float*)   alloc((size_t)SEQ * KVH * HD * 4);
  _Float16* qh   = (_Float16*)alloc((size_t)SEQ * NH * HD * 2);
  _Float16* khb  = (_Float16*)alloc((size_t)SEQ * KVH * HD * 2);
  _Float16* vpk  = (_Float16*)alloc((size_t)SEQ * KVH * HD * 2);
  _Float16* ohb  = (_Float16*)alloc((size_t)SEQ * NH * HD * 2);
  float*    x1   = (float*)   alloc((size_t)SEQ * DIM * 4);
  float*    h2f  = (float*)   alloc((size_t)SEQ * DIM * 4);
  _Float16* h2h  = (_Float16*)alloc((size_t)SEQ * DIM * 2);
  float*    gatew   = (float*)alloc((size_t)SEQ * NEXP * 4);
  int*      counts  = (int*)  alloc(NEXP * 4);
  int*      offsets = (int*)  alloc((NEXP + 1) * 4);
  int*      bucket  = (int*)  alloc((size_t)NEXP * SEQ * 4);
  _Float16* actb = (_Float16*)alloc((size_t)SEQ * TOPK * HID * 2);
  float*    moey = (float*)   alloc((size_t)SEQ * TOPK * DIM * 4);
  _Float16* wqp = (_Float16*)alloc((size_t)DIM * NH * HD * 2);
  _Float16* wkp = (_Float16*)alloc((size_t)DIM * KVH * HD * 2);
  _Float16* wvp = (_Float16*)alloc((size_t)DIM * KVH * HD * 2);
  _Float16* wop = (_Float16*)alloc((size_t)NH * HD * DIM * 2);
  _Float16* w1p = (_Float16*)alloc((size_t)NEXP * DIM * HID * 2);
  _Float16* w3p = (_Float16*)alloc((size_t)NEXP * DIM * HID * 2);
  _Float16* w2p = (_Float16*)alloc((size_t)NEXP * HID * DIM * 2);
  (void)ws_size; (void)in_sizes; (void)n_in; (void)out_size;

  const dim3 gwave(32, 4);
  auto pack_blocks = [](int K, int N) { return (K * N / 8 + 255) / 256; };

  // 0) pre-pack all weights into WMMA-B fragment order (one bandwidth pass)
  k_pack<<<dim3(pack_blocks(DIM, NH * HD), 1), 256, 0, stream>>>(wq, wqp, DIM, NH * HD);
  k_pack<<<dim3(pack_blocks(DIM, KVH * HD), 1), 256, 0, stream>>>(wk, wkp, DIM, KVH * HD);
  k_pack<<<dim3(pack_blocks(DIM, KVH * HD), 1), 256, 0, stream>>>(wv, wvp, DIM, KVH * HD);
  k_pack<<<dim3(pack_blocks(NH * HD, DIM), 1), 256, 0, stream>>>(wo, wop, NH * HD, DIM);
  k_pack<<<dim3(pack_blocks(DIM, HID), NEXP), 256, 0, stream>>>(w1, w1p, DIM, HID);
  k_pack<<<dim3(pack_blocks(DIM, HID), NEXP), 256, 0, stream>>>(w3, w3p, DIM, HID);
  k_pack<<<dim3(pack_blocks(HID, DIM), NEXP), 256, 0, stream>>>(w2, w2p, HID, DIM);

  // 1) h1 = rms(x) * ln_w
  k_rmsnorm<<<SEQ, 256, 0, stream>>>(x, ln_w, nullptr, h1h);
  // 2) QKV projections (WMMA, TDM-staged A, packed B)
  k_gemm<<<dim3(SEQ / 16, (NH * HD) / 256), gwave, 0, stream>>>(h1h, wqp, nullptr, qf, nullptr, SEQ, NH * HD, DIM);
  k_gemm<<<dim3(SEQ / 16, (KVH * HD) / 256), gwave, 0, stream>>>(h1h, wkp, nullptr, kf, nullptr, SEQ, KVH * HD, DIM);
  k_gemm<<<dim3(SEQ / 16, (KVH * HD) / 256), gwave, 0, stream>>>(h1h, wvp, nullptr, vf, nullptr, SEQ, KVH * HD, DIM);
  // 3) RoPE + V packing
  {
    int tq = SEQ * NH * 32, tk = SEQ * KVH * 32, tv = SEQ * KVH * HD;
    k_rope<<<(tq + 255) / 256, 256, 0, stream>>>(qf, qh, NH, tq);
    k_rope<<<(tk + 255) / 256, 256, 0, stream>>>(kf, khb, KVH, tk);
    k_pack_v<<<(tv + 255) / 256, 256, 0, stream>>>(vf, vpk);
  }
  // 4) Flash attention (WMMA)
  k_attention<<<dim3(SEQ / 16, NH), 32, 0, stream>>>(qh, khb, vpk, ohb);
  // 5) x1 = x + attn @ wo
  k_gemm<<<dim3(SEQ / 16, DIM / 256), gwave, 0, stream>>>(ohb, wop, x, x1, nullptr, SEQ, DIM, NH * HD);
  // 6) h2 = rms(x1) * ff_ln_w
  k_rmsnorm<<<SEQ, 256, 0, stream>>>(x1, ff_w, h2f, h2h);
  // 7) gate + top-4 routing
  k_zero_counts<<<1, 32, 0, stream>>>(counts);
  k_gate<<<SEQ, 256, 0, stream>>>(h2f, wg, gatew, counts, bucket);
  k_scan<<<1, 1, 0, stream>>>(counts, offsets);
  // 8) expert GEMMs (routed tokens only, WMMA, packed B)
  k_moe1<<<dim3(NEXP, SEQ / 16, HID / 256), gwave, 0, stream>>>(h2h, w1p, w3p, counts, offsets, bucket, actb);
  k_moe2<<<dim3(NEXP, SEQ / 16, DIM / 256), gwave, 0, stream>>>(actb, w2p, gatew, counts, offsets, bucket, moey);
  // 9) out = x1 + sum of weighted expert outputs
  k_combine<<<(SEQ * DIM + 255) / 256, 256, 0, stream>>>(x1, moey, out);
}
